// WhisperAttention_87325275062661
// MI455X (gfx1250) — matrix operand order repaired and checked by reading, not compiled, not run on hardware
//
#include <hip/hip_runtime.h>
#include <hip/hip_bf16.h>

// ---------------------------------------------------------------------------
// Whisper decoder self-attention for MI455X (gfx1250, wave32, WMMA + TDM).
// B=2, S=2048, E=1280, H=20, D=64.
// ---------------------------------------------------------------------------

#define B_   2
#define S_   2048
#define E_   1280
#define H_   20
#define D_   64
#define N3E_ 3840
#define M_   (B_ * S_)        // 4096
#define SCALING 0.125f        // 64^-0.5
#define LOG2E 1.4426950408889634f

typedef _Float16 h16 __attribute__((ext_vector_type(16)));
typedef _Float16 h8  __attribute__((ext_vector_type(8)));
typedef _Float16 h4  __attribute__((ext_vector_type(4)));
typedef float    f8  __attribute__((ext_vector_type(8)));
typedef float    f32x4 __attribute__((ext_vector_type(4)));
typedef unsigned int u32x4 __attribute__((ext_vector_type(4)));
typedef int      i32x4 __attribute__((ext_vector_type(4)));
typedef int      i32x8 __attribute__((ext_vector_type(8)));

// ---- WMMA fragment loaders (contiguous k = 16*hid + s, same for A and B) ------
__device__ __forceinline__ h16 load_a_frag(const _Float16* row, int hid) {
  h16 r;
  const int kb = hid * 16;
  r.lo = *(const h8*)(row + kb);
  r.hi = *(const h8*)(row + kb + 8);
  return r;
}
__device__ __forceinline__ h16 load_b_frag(const _Float16* colp, int hid) {
  h16 r;
  const _Float16* p = colp + hid * 16;
  r.lo = *(const h8*)(p);
  r.hi = *(const h8*)(p + 8);
  return r;
}

#define WMMA_F16(a, b, c) \
  __builtin_amdgcn_wmma_f32_16x16x32_f16(false, (a), false, (b), (short)0, (c), false, false)

// ---- Tensor Data Mover: 2D f16 tile (rows x cols) global -> LDS ------------
// D# per CDNA5 ISA.  pad_* in encoded form (0 = off).
__device__ __forceinline__ void tdm_load_2d_f16(unsigned lds_off, const void* gaddr,
                                                int rows, int cols, int tensor_rows,
                                                int row_stride_elems,
                                                int pad_en, int pad_ival, int pad_amt) {
  const unsigned long long ga = (unsigned long long)(uintptr_t)gaddr;
  u32x4 g0;
  g0[0] = 1u;                                               // count=1 user D#
  g0[1] = lds_off;                                          // lds_addr (bytes)
  g0[2] = (unsigned)(ga & 0xFFFFFFFFu);                     // global_addr lo
  g0[3] = (unsigned)((ga >> 32) & 0x01FFFFFFu) | (2u << 30);// addr hi | type=2
  i32x8 g1;
  g1[0] = (1 << 16) | (pad_en << 20) | (pad_ival << 22) | (pad_amt << 25); // 2B elems
  g1[1] = (int)((unsigned)cols << 16);                      // tensor_dim0[15:0]
  g1[2] = (int)(((unsigned)cols >> 16) | ((unsigned)tensor_rows << 16));
  g1[3] = (int)(((unsigned)tensor_rows >> 16) | ((unsigned)cols << 16)); // tile_dim0
  g1[4] = rows & 0xFFFF;                                    // tile_dim1 (tile_dim2=0)
  g1[5] = row_stride_elems;                                 // dim0_stride lo32
  g1[6] = 0;
  g1[7] = 0;
  i32x4 z4 = {0, 0, 0, 0};
#if defined(__clang_major__) && __clang_major__ >= 23
  i32x8 z8 = {0, 0, 0, 0, 0, 0, 0, 0};
  __builtin_amdgcn_tensor_load_to_lds(g0, g1, z4, z4, z8, 0);
#else
  __builtin_amdgcn_tensor_load_to_lds(g0, g1, z4, z4, 0);
#endif
}

// ---------------------------------------------------------------------------
// Kernel 1: fused QKV projection.  X[4096,1280] @ W[1280,3840] + bias.
// (fp32 -> f16 conversion during staging, so no TDM here.)
// ---------------------------------------------------------------------------
__global__ __launch_bounds__(256) void qkv_gemm_kernel(
    const float* __restrict__ X, const float* __restrict__ W,
    const float* __restrict__ bias,
    _Float16* __restrict__ Qb, _Float16* __restrict__ Kb, _Float16* __restrict__ Vb) {
  __shared__ __align__(16) _Float16 sA[128][40];   // [m][k]
  __shared__ __align__(16) _Float16 sB[128][40];   // [n][k] (transposed)

  const int tid  = threadIdx.x;
  const int lane = tid & 31, wave = tid >> 5;
  const int col = lane & 15, hid = lane >> 4;
  const int wm = (wave >> 2) * 64;
  const int wn = (wave & 3) * 32;
  const int m0 = blockIdx.y * 128;
  const int n0 = blockIdx.x * 128;

  f8 acc[4][2] = {};

  for (int k0 = 0; k0 < E_; k0 += 32) {
    __syncthreads();
#pragma unroll
    for (int j = 0; j < 4; ++j) {                    // A tile 128x32
      const int c = tid + j * 256;
      const int m = c >> 3, k4 = (c & 7) << 2;
      f32x4 v = *(const f32x4*)(X + (size_t)(m0 + m) * E_ + k0 + k4);
      h4 hv = {(_Float16)v.x, (_Float16)v.y, (_Float16)v.z, (_Float16)v.w};
      *(h4*)(&sA[m][k4]) = hv;
    }
#pragma unroll
    for (int j = 0; j < 4; ++j) {                    // B tile 32x128 -> [n][k]
      const int c = tid + j * 256;
      const int k = c >> 5, n4 = (c & 31) << 2;
      f32x4 v = *(const f32x4*)(W + (size_t)(k0 + k) * N3E_ + n0 + n4);
      sB[n4 + 0][k] = (_Float16)v.x;
      sB[n4 + 1][k] = (_Float16)v.y;
      sB[n4 + 2][k] = (_Float16)v.z;
      sB[n4 + 3][k] = (_Float16)v.w;
    }
    if (k0 + 32 < E_) {
      __builtin_prefetch(X + (size_t)(m0 + (tid & 127)) * E_ + k0 + 32, 0, 1);
      __builtin_prefetch(W + (size_t)(k0 + 32 + (tid & 31)) * N3E_ + n0, 0, 1);
    }
    __syncthreads();

    h16 af[4], bf[2];
#pragma unroll
    for (int mt = 0; mt < 4; ++mt) af[mt] = load_a_frag(&sA[wm + mt * 16 + col][0], hid);
#pragma unroll
    for (int nt = 0; nt < 2; ++nt) bf[nt] = load_b_frag(&sB[wn + nt * 16 + col][0], hid);
#pragma unroll
    for (int mt = 0; mt < 4; ++mt)
#pragma unroll
      for (int nt = 0; nt < 2; ++nt)
        acc[mt][nt] = WMMA_F16(af[mt], bf[nt], acc[mt][nt]);
  }

#pragma unroll
  for (int mt = 0; mt < 4; ++mt)
#pragma unroll
    for (int nt = 0; nt < 2; ++nt)
#pragma unroll
      for (int i = 0; i < 8; ++i) {
        const int m = m0 + wm + mt * 16 + i + 8 * hid;
        const int n = n0 + wn + nt * 16 + col;
        float v = acc[mt][nt][i] + bias[n];
        const int part = n / E_;
        const int nn = n - part * E_;
        const int h = nn >> 6, d = nn & 63;
        const int b = m >> 11, s = m & (S_ - 1);
        const size_t idx = ((size_t)(b * H_ + h) * S_ + s) * D_ + d;
        if (part == 0)      Qb[idx] = (_Float16)(v * SCALING);
        else if (part == 1) Kb[idx] = (_Float16)v;
        else                Vb[idx] = (_Float16)v;
      }
}

// ---------------------------------------------------------------------------
// Kernel 2: causal flash attention.  64 q rows per block, 4 waves x 16 rows.
// K tile staged by the Tensor Data Mover (overlapped with the manual V
// transpose fill); online softmax; P@V on WMMA.
// ---------------------------------------------------------------------------
__device__ __forceinline__ float rowmax16(float v) {
#pragma unroll
  for (int off = 8; off; off >>= 1) v = fmaxf(v, __shfl_xor(v, off, 32));
  return v;
}
__device__ __forceinline__ float rowsum16(float v) {
#pragma unroll
  for (int off = 8; off; off >>= 1) v += __shfl_xor(v, off, 32);
  return v;
}

__global__ __launch_bounds__(128) void attn_kernel(
    const _Float16* __restrict__ Qb, const _Float16* __restrict__ Kb,
    const _Float16* __restrict__ Vb, _Float16* __restrict__ attn) {
  __shared__ __align__(16) _Float16 sK[32][64];       // [key][d] (TDM target)
  __shared__ __align__(16) _Float16 sVT[64][40];      // [d][key]
  __shared__ __align__(16) _Float16 sP[4][16][40];    // per-wave P staging

  const int tid  = threadIdx.x;
  const int lane = tid & 31, wave = tid >> 5;
  const int col = lane & 15, hid = lane >> 4;
  const int bh = blockIdx.y;
  const int b = bh / H_, h = bh % H_;
  const int qbase = blockIdx.x * 64;

  const _Float16* Qh = Qb + (size_t)bh * S_ * D_;
  const _Float16* Kh = Kb + (size_t)bh * S_ * D_;
  const _Float16* Vh = Vb + (size_t)bh * S_ * D_;

  const int qrow = qbase + wave * 16 + col;
  h16 qf[2];
  qf[0] = load_a_frag(Qh + (size_t)qrow * D_,      hid);
  qf[1] = load_a_frag(Qh + (size_t)qrow * D_ + 32, hid);

  f8 o[4] = {};
  float mrow[8], lrow[8];
#pragma unroll
  for (int i = 0; i < 8; ++i) { mrow[i] = -1e30f; lrow[i] = 0.0f; }

  const int ntiles = (qbase + 64) / 32;
  for (int t = 0; t < ntiles; ++t) {
    const int kt = t * 32;
    __syncthreads();
    if (wave == 0)                                    // K tile: async DMA
      tdm_load_2d_f16((unsigned)(uintptr_t)&sK[0][0], Kh + (size_t)kt * D_,
                      /*rows=*/32, /*cols=*/D_, /*tensor_rows=*/S_, /*stride=*/D_,
                      0, 0, 0);
    // V tile transposed [d][key], overlapped with the DMA.
#pragma unroll
    for (int j = 0; j < 2; ++j) {
      const int c = tid + j * 128;
      const int r = c >> 3, d8 = (c & 7) << 3;
      h8 v = *(const h8*)(Vh + (size_t)(kt + r) * D_ + d8);
#pragma unroll
      for (int e = 0; e < 8; ++e) sVT[d8 + e][r] = v[e];
    }
    if (wave == 0) __builtin_amdgcn_s_wait_tensorcnt(0);
    __syncthreads();

    // scores: 16 q rows x 32 keys, contraction over D=64.
    f8 s0 = {}, s1 = {};
#pragma unroll
    for (int dh = 0; dh < 2; ++dh) {
      h16 bk0 = load_b_frag(&sK[col][dh * 32],      hid);
      h16 bk1 = load_b_frag(&sK[col + 16][dh * 32], hid);
      s0 = WMMA_F16(qf[dh], bk0, s0);
      s1 = WMMA_F16(qf[dh], bk1, s1);
    }

    // online softmax (rows i+8*hid, cols lane&15 / +16).
#pragma unroll
    for (int i = 0; i < 8; ++i) {
      const int qi = qbase + wave * 16 + i + 8 * hid;
      const float v0 = (kt + col      <= qi) ? s0[i] : -1e30f;
      const float v1 = (kt + 16 + col <= qi) ? s1[i] : -1e30f;
      const float tmax = rowmax16(fmaxf(v0, v1));
      const float nm = fmaxf(mrow[i], tmax);
      const float al = exp2f((mrow[i] - nm) * LOG2E);
      const float e0 = exp2f((v0 - nm) * LOG2E);
      const float e1 = exp2f((v1 - nm) * LOG2E);
      lrow[i] = lrow[i] * al + rowsum16(e0 + e1);
      mrow[i] = nm;
#pragma unroll
      for (int dt = 0; dt < 4; ++dt) o[dt][i] *= al;
      sP[wave][i + 8 * hid][col]      = (_Float16)e0;
      sP[wave][i + 8 * hid][col + 16] = (_Float16)e1;
    }
    __syncthreads();

    // P @ V : A = P (16x32), B = V^T columns (contraction over 32 keys).
    h16 pf = load_a_frag(&sP[wave][col][0], hid);
#pragma unroll
    for (int dt = 0; dt < 4; ++dt) {
      h16 bv = load_b_frag(&sVT[dt * 16 + col][0], hid);
      o[dt] = WMMA_F16(pf, bv, o[dt]);
    }
  }

#pragma unroll
  for (int i = 0; i < 8; ++i) {
    const float rl = 1.0f / lrow[i];
    const int q = qbase + wave * 16 + i + 8 * hid;
#pragma unroll
    for (int dt = 0; dt < 4; ++dt) {
      const int d = dt * 16 + col;
      attn[((size_t)(b * S_ + q)) * E_ + h * D_ + d] = (_Float16)(o[dt][i] * rl);
    }
  }
}

// ---------------------------------------------------------------------------
// Kernel 3: output projection. attn[4096,1280](f16) @ out_w[1280,1280] + b.
// A tile staged by TDM with LDS padding (interval 16 DW, amount 4 DW -> the
// [128][40] layout), overlapped with the fp32->f16 B-tile transpose fill.
// ---------------------------------------------------------------------------
__global__ __launch_bounds__(256) void out_gemm_kernel(
    const _Float16* __restrict__ A, const float* __restrict__ W,
    const float* __restrict__ bias, float* __restrict__ out) {
  __shared__ __align__(16) _Float16 sA[128][40];
  __shared__ __align__(16) _Float16 sB[128][40];

  const int tid  = threadIdx.x;
  const int lane = tid & 31, wave = tid >> 5;
  const int col = lane & 15, hid = lane >> 4;
  const int wm = (wave >> 2) * 64;
  const int wn = (wave & 3) * 32;
  const int m0 = blockIdx.y * 128;
  const int n0 = blockIdx.x * 128;

  f8 acc[4][2] = {};

  for (int k0 = 0; k0 < E_; k0 += 32) {
    __syncthreads();
    if (wave == 0)                                    // A tile: async DMA + pad
      tdm_load_2d_f16((unsigned)(uintptr_t)&sA[0][0],
                      A + (size_t)m0 * E_ + k0,
                      /*rows=*/128, /*cols=*/32, /*tensor_rows=*/M_, /*stride=*/E_,
                      /*pad_en=*/1, /*pad_ival=*/3 /*16 DW*/, /*pad_amt=*/3 /*4 DW*/);
#pragma unroll
    for (int j = 0; j < 4; ++j) {                     // B tile 32x128 -> [n][k]
      const int c = tid + j * 256;
      const int k = c >> 5, n4 = (c & 31) << 2;
      f32x4 v = *(const f32x4*)(W + (size_t)(k0 + k) * E_ + n0 + n4);
      sB[n4 + 0][k] = (_Float16)v.x;
      sB[n4 + 1][k] = (_Float16)v.y;
      sB[n4 + 2][k] = (_Float16)v.z;
      sB[n4 + 3][k] = (_Float16)v.w;
    }
    if (k0 + 32 < E_)
      __builtin_prefetch(W + (size_t)(k0 + 32 + (tid & 31)) * E_ + n0, 0, 1);
    if (wave == 0) __builtin_amdgcn_s_wait_tensorcnt(0);
    __syncthreads();

    h16 af[4], bf[2];
#pragma unroll
    for (int mt = 0; mt < 4; ++mt) af[mt] = load_a_frag(&sA[wm + mt * 16 + col][0], hid);
#pragma unroll
    for (int nt = 0; nt < 2; ++nt) bf[nt] = load_b_frag(&sB[wn + nt * 16 + col][0], hid);
#pragma unroll
    for (int mt = 0; mt < 4; ++mt)
#pragma unroll
      for (int nt = 0; nt < 2; ++nt)
        acc[mt][nt] = WMMA_F16(af[mt], bf[nt], acc[mt][nt]);
  }

#pragma unroll
  for (int mt = 0; mt < 4; ++mt)
#pragma unroll
    for (int nt = 0; nt < 2; ++nt)
#pragma unroll
      for (int i = 0; i < 8; ++i) {
        const int m = m0 + wm + mt * 16 + i + 8 * hid;
        const int n = n0 + wn + nt * 16 + col;
        out[(size_t)m * E_ + n] = acc[mt][nt][i] + bias[n];
      }
}

// ---------------------------------------------------------------------------
extern "C" void kernel_launch(void* const* d_in, const int* in_sizes, int n_in,
                              void* d_out, int out_size, void* d_ws, size_t ws_size,
                              hipStream_t stream) {
  const float* hs    = (const float*)d_in[0];
  const float* qkv_w = (const float*)d_in[1];
  const float* qkv_b = (const float*)d_in[2];
  const float* out_w = (const float*)d_in[3];
  const float* out_b = (const float*)d_in[4];
  float* out = (float*)d_out;

  const size_t HBUF = (size_t)B_ * H_ * S_ * D_;  // 5,242,880 halves
  _Float16* ws   = (_Float16*)d_ws;
  _Float16* Qb   = ws;
  _Float16* Kb   = ws + HBUF;
  _Float16* Vb   = ws + 2 * HBUF;
  _Float16* attn = ws + 3 * HBUF;

  dim3 g1(N3E_ / 128, M_ / 128);
  qkv_gemm_kernel<<<g1, 256, 0, stream>>>(hs, qkv_w, qkv_b, Qb, Kb, Vb);

  dim3 g2(S_ / 64, B_ * H_);
  attn_kernel<<<g2, 128, 0, stream>>>(Qb, Kb, Vb, attn);

  dim3 g3(E_ / 128, M_ / 128);
  out_gemm_kernel<<<g3, 256, 0, stream>>>(attn, out_w, out_b, out);
}
